// SynthesisNet_60232621359216
// MI455X (gfx1250) — matrix operand order repaired and checked by reading, not compiled, not run on hardware
//
#include <hip/hip_runtime.h>
#include <hip/hip_bf16.h>

typedef __bf16 bf16_t;
typedef __attribute__((ext_vector_type(16))) __bf16 v16bf;
typedef __attribute__((ext_vector_type(8)))  __bf16 v8bf;
typedef __attribute__((ext_vector_type(8)))  float   v8f;

// Problem constants
constexpr int Bc = 4096, Sc = 512, Kc = 10, Mc = 20, Uc = 256, Cc = 128;
constexpr int Rc = Bc * 5;                 // MDN rows
constexpr int KP1 = 672;                   // 3+128+512=643 -> pad 672
constexpr int KP2 = 1184;                  // 3+512+128+512=1155 -> pad 1184
constexpr int KPO = 1024;                  // 512+512
constexpr int KPH = 512;
constexpr int NWIN = 64;                   // 60 -> pad 64
constexpr int NLIN = 128;                  // 121 -> pad 128

// Output layout (floats), reference return order:
// (eos, pi, mu1, mu2, sigma1, sigma2, rho, h1, h2, osc_new, w, phi, kappa)
constexpr size_t OFF_EOS   = 0;
constexpr size_t OFF_PI    = OFF_EOS + (size_t)Rc;
constexpr size_t OFF_MU1   = OFF_PI  + (size_t)Rc * Mc;
constexpr size_t OFF_MU2   = OFF_MU1 + (size_t)Rc * Mc;
constexpr size_t OFF_S1    = OFF_MU2 + (size_t)Rc * Mc;
constexpr size_t OFF_S2    = OFF_S1  + (size_t)Rc * Mc;
constexpr size_t OFF_RHO   = OFF_S2  + (size_t)Rc * Mc;
constexpr size_t OFF_H1    = OFF_RHO + (size_t)Rc * Mc;
constexpr size_t OFF_H2    = OFF_H1  + (size_t)Bc * Sc;
constexpr size_t OFF_OSC   = OFF_H2  + (size_t)Bc * Sc;
constexpr size_t OFF_W     = OFF_OSC + 4ull * Bc * Sc;
constexpr size_t OFF_PHI   = OFF_W   + (size_t)Bc * Cc;
constexpr size_t OFF_KAPPA = OFF_PHI + (size_t)Bc * Uc;

// ---------------------------------------------------------------------------
// Pack [Ka,N]+[Kb,N] f32 weights into transposed, zero-padded bf16 [Npad, Kp].
// grid: (ceil(Kp/256), Npad)
// ---------------------------------------------------------------------------
__global__ void pack_wt(bf16_t* __restrict__ dst, int Kp, int Nsrc,
                        const float* __restrict__ A, int Ka,
                        const float* __restrict__ Bsrc, int Kb) {
  int n = blockIdx.y;
  int k = blockIdx.x * blockDim.x + threadIdx.x;
  if (k >= Kp) return;
  float v = 0.f;
  if (n < Nsrc) {
    if (k < Ka)                    v = A[(size_t)k * Nsrc + n];
    else if (Bsrc && k < Ka + Kb)  v = Bsrc[(size_t)(k - Ka) * Nsrc + n];
  }
  dst[(size_t)n * Kp + k] = (bf16_t)v;
}

__global__ void pack_bias(float* __restrict__ dst,
                          const float* __restrict__ a,
                          const float* __restrict__ b, int Nsrc, int Npad) {
  int n = blockIdx.x * blockDim.x + threadIdx.x;
  if (n >= Npad) return;
  float v = 0.f;
  if (n < Nsrc) { v = a[n]; if (b) v += b[n]; }
  dst[n] = v;
}

// ---------------------------------------------------------------------------
// Concatenate up to 4 f32 row-major sources into one bf16 [rows, Kp] matrix.
// grid: (ceil(Kp/256), rows)
// ---------------------------------------------------------------------------
__global__ void concat_bf16(bf16_t* __restrict__ dst, int Kp,
                            const float* __restrict__ s0, int w0,
                            const float* __restrict__ s1, int w1,
                            const float* __restrict__ s2, int w2,
                            const float* __restrict__ s3, int w3) {
  int row = blockIdx.y;
  int k = blockIdx.x * blockDim.x + threadIdx.x;
  if (k >= Kp) return;
  float v = 0.f;
  int off = k;
  if (off < w0) v = s0[(size_t)row * w0 + off];
  else {
    off -= w0;
    if (off < w1) v = s1[(size_t)row * w1 + off];
    else {
      off -= w1;
      if (off < w2) v = s2[(size_t)row * w2 + off];
      else {
        off -= w2;
        if (s3 && off < w3) v = s3[(size_t)row * w3 + off];
      }
    }
  }
  dst[(size_t)row * Kp + k] = (bf16_t)v;
}

// ---------------------------------------------------------------------------
// BF16 WMMA GEMM: C[M,Npad] = A[M,Kp] * BT[Npad,Kp]^T + bias.
// One wave computes a 32(M) x 64(N) tile: 2 A-frags x 4 B-frags -> 8 WMMAs
// per K-step; every B-fragment is reused twice (halves B-side VMEM traffic).
// Fragment layouts: A and B both use contiguous K order (k = 16h + s), not the ISA table order.
// block=128 (4 waves); grid.x = total_tiles/4 (exact multiples by construction).
// ---------------------------------------------------------------------------
__global__ __launch_bounds__(128)
void gemm_bf16_wmma(const bf16_t* __restrict__ A,
                    const bf16_t* __restrict__ BT,
                    const float*  __restrict__ bias,
                    float* __restrict__ Cm,
                    int M, int Npad, int Kp) {
  const int lane = threadIdx.x & 31;
  const int wave = threadIdx.x >> 5;
  const int tiles_n = Npad >> 6;
  const int tiles_m = M >> 5;
  const int tile = blockIdx.x * 4 + wave;
  if (tile >= tiles_m * tiles_n) return;
  const int m0 = (tile / tiles_n) << 5;
  const int n0 = (tile % tiles_n) << 6;
  const int h = lane >> 4;   // lane half selects K[16h .. 16h+15] in both A and B
  const int l = lane & 15;

  const bf16_t* aRow0 = A  + (size_t)(m0 + l) * Kp;
  const bf16_t* aRow1 = A  + (size_t)(m0 + 16 + l) * Kp;
  const bf16_t* bRow0 = BT + (size_t)(n0 + l) * Kp + 16 * h;
  const size_t  bstr  = (size_t)16 * Kp;

  v8f acc00 = {}; v8f acc01 = {}; v8f acc02 = {}; v8f acc03 = {};
  v8f acc10 = {}; v8f acc11 = {}; v8f acc12 = {}; v8f acc13 = {};
  for (int k = 0; k < Kp; k += 32) {
    // A 16x32: lane half h -> elems = K[k+16h .. k+16h+15] contiguous
    v16bf a0 = *(const v16bf*)(aRow0 + k + 16 * h);
    v16bf a1 = *(const v16bf*)(aRow1 + k + 16 * h);
    // B 32x16: lane n = col, elems = K[k+16h .. k+16h+15] contiguous in BT row
    v16bf b0 = *(const v16bf*)(bRow0 + k);
    v16bf b1 = *(const v16bf*)(bRow0 + bstr + k);
    v16bf b2 = *(const v16bf*)(bRow0 + 2 * bstr + k);
    v16bf b3 = *(const v16bf*)(bRow0 + 3 * bstr + k);
    acc00 = __builtin_amdgcn_wmma_f32_16x16x32_bf16(false, a0, false, b0, (short)0, acc00, false, false);
    acc10 = __builtin_amdgcn_wmma_f32_16x16x32_bf16(false, a1, false, b0, (short)0, acc10, false, false);
    acc01 = __builtin_amdgcn_wmma_f32_16x16x32_bf16(false, a0, false, b1, (short)0, acc01, false, false);
    acc11 = __builtin_amdgcn_wmma_f32_16x16x32_bf16(false, a1, false, b1, (short)0, acc11, false, false);
    acc02 = __builtin_amdgcn_wmma_f32_16x16x32_bf16(false, a0, false, b2, (short)0, acc02, false, false);
    acc12 = __builtin_amdgcn_wmma_f32_16x16x32_bf16(false, a1, false, b2, (short)0, acc12, false, false);
    acc03 = __builtin_amdgcn_wmma_f32_16x16x32_bf16(false, a0, false, b3, (short)0, acc03, false, false);
    acc13 = __builtin_amdgcn_wmma_f32_16x16x32_bf16(false, a1, false, b3, (short)0, acc13, false, false);
  }
  const float bi0 = bias[n0 + l];
  const float bi1 = bias[n0 + 16 + l];
  const float bi2 = bias[n0 + 32 + l];
  const float bi3 = bias[n0 + 48 + l];
#pragma unroll
  for (int r = 0; r < 8; ++r) {
    // C/D 16x16 f32: VGPR r, lane -> row m0+r+8h, col n0+l (+16 per sub-tile)
    float* crow0 = Cm + (size_t)(m0 + r + 8 * h) * Npad + n0 + l;
    crow0[0]  = acc00[r] + bi0;
    crow0[16] = acc01[r] + bi1;
    crow0[32] = acc02[r] + bi2;
    crow0[48] = acc03[r] + bi3;
    float* crow1 = Cm + (size_t)(m0 + 16 + r + 8 * h) * Npad + n0 + l;
    crow1[0]  = acc10[r] + bi0;
    crow1[16] = acc11[r] + bi1;
    crow1[32] = acc12[r] + bi2;
    crow1[48] = acc13[r] + bi3;
  }
}

// ---------------------------------------------------------------------------
// FF gated cell epilogue: h' = sig(pre_j)*(1-h) + (1-sig(pre_k))*h
// ---------------------------------------------------------------------------
__global__ void gate_combine(const float* __restrict__ prej,
                             const float* __restrict__ prek,
                             const float* __restrict__ h,
                             float* __restrict__ hnew,
                             bf16_t* __restrict__ hnew_bf, int n) {
  int i = blockIdx.x * blockDim.x + threadIdx.x;
  if (i >= n) return;
  float j  = 1.f / (1.f + __expf(-prej[i]));
  float kk = 1.f / (1.f + __expf(-prek[i]));
  float hv = h[i];
  float o  = j * (1.f - hv) + (1.f - kk) * hv;
  hnew[i] = o;
  if (hnew_bf) hnew_bf[i] = (bf16_t)o;
}

// ---------------------------------------------------------------------------
// Gaussian window: kg[B,64] (cols 0..29 used) -> phi[B,U], kappa[B,K]
// grid: B blocks of 256 threads (thread = u)
// ---------------------------------------------------------------------------
__global__ void window_kernel(const float* __restrict__ kg,
                              const float* __restrict__ init_kappa,
                              float* __restrict__ phi_out,
                              float* __restrict__ kappa_out) {
  int b = blockIdx.x;
  int t = threadIdx.x;
  __shared__ float al[Kc], be[Kc], ka[Kc];
  if (t < Kc) {
    al[t] = __expf(kg[(size_t)b * NWIN + t]);
    be[t] = __expf(kg[(size_t)b * NWIN + Kc + t]);
    float kv = init_kappa[(size_t)b * Kc + t] + __expf(kg[(size_t)b * NWIN + 2 * Kc + t]);
    ka[t] = kv;
    kappa_out[(size_t)b * Kc + t] = kv;
  }
  __syncthreads();
  float u = (float)t;
  float s = 0.f;
#pragma unroll
  for (int k = 0; k < Kc; ++k) {
    float d = ka[k] - u;
    s += __expf(-be[k] * d * d) * al[k];
  }
  phi_out[(size_t)b * Uc + t] = s;
}

// ---------------------------------------------------------------------------
// w[b,c] = sum_u phi[b,u] * c_vec[b,u,c]   grid: B blocks of 128 threads
// ---------------------------------------------------------------------------
__global__ void attend_kernel(const float* __restrict__ phi,
                              const float* __restrict__ cvec,
                              float* __restrict__ w) {
  int b = blockIdx.x, c = threadIdx.x;
  __shared__ float p[Uc];
  p[c]        = phi[(size_t)b * Uc + c];
  p[c + 128]  = phi[(size_t)b * Uc + c + 128];
  __syncthreads();
  const float* cv = cvec + (size_t)b * Uc * Cc + c;
  float s = 0.f;
#pragma unroll 4
  for (int u = 0; u < Uc; ++u) s += p[u] * cv[(size_t)u * Cc];
  w[(size_t)b * Cc + c] = s;
}

// ---------------------------------------------------------------------------
// Oscillator recurrence (5 steps) -> bf16 osc_out rows [B*5, S] for MDN GEMM
// ---------------------------------------------------------------------------
__global__ void osc_recur(const float* __restrict__ osc_new,
                          bf16_t* __restrict__ osc_out) {
  int i = blockIdx.x * blockDim.x + threadIdx.x;
  if (i >= Bc * Sc) return;
  const size_t BS = (size_t)Bc * Sc;
  float ph = osc_new[0 * BS + i];
  float om = osc_new[1 * BS + i];
  float r  = osc_new[2 * BS + i];
  float mu = osc_new[3 * BS + i];
  int b = i >> 9, s = i & (Sc - 1);
  bf16_t* orow = osc_out + (size_t)b * 5 * Sc + s;
  orow[0] = (bf16_t)(r * __cosf(ph));
#pragma unroll
  for (int t = 1; t < 5; ++t) {
    r  = r + (mu - r * r) * r;
    ph = ph + om;
    orow[(size_t)t * Sc] = (bf16_t)(r * __cosf(ph));
  }
}

// ---------------------------------------------------------------------------
// MDN head epilogue: one thread per row of out[R,128] (cols 0..120 valid)
// ---------------------------------------------------------------------------
__global__ void mdn_epilogue(const float* __restrict__ o, float* __restrict__ d, int R) {
  int r = blockIdx.x * blockDim.x + threadIdx.x;
  if (r >= R) return;
  const float* row = o + (size_t)r * NLIN;
  d[OFF_EOS + r] = 1.f / (1.f + __expf(row[0]));
  float mx = -3.4e38f;
#pragma unroll
  for (int i = 0; i < Mc; ++i) mx = fmaxf(mx, row[1 + i]);
  float e[Mc]; float den = 0.f;
#pragma unroll
  for (int i = 0; i < Mc; ++i) { e[i] = __expf(row[1 + i] - mx); den += e[i]; }
  float inv = 1.f / den;
#pragma unroll
  for (int i = 0; i < Mc; ++i) {
    d[OFF_PI  + (size_t)r * Mc + i] = e[i] * inv;
    d[OFF_MU1 + (size_t)r * Mc + i] = row[1 + Mc + i];
    d[OFF_MU2 + (size_t)r * Mc + i] = row[1 + 2 * Mc + i];
    d[OFF_S1  + (size_t)r * Mc + i] = __expf(row[1 + 3 * Mc + i]);
    d[OFF_S2  + (size_t)r * Mc + i] = __expf(row[1 + 4 * Mc + i]);
    d[OFF_RHO + (size_t)r * Mc + i] = tanhf(row[1 + 5 * Mc + i]);
  }
}

// ===========================================================================
extern "C" void kernel_launch(void* const* d_in, const int* in_sizes, int n_in,
                              void* d_out, int out_size, void* d_ws, size_t ws_size,
                              hipStream_t stream) {
  (void)in_sizes; (void)n_in; (void)out_size; (void)ws_size;
  const float* x          = (const float*)d_in[0];
  const float* c_vec      = (const float*)d_in[1];
  const float* rnn1_h     = (const float*)d_in[2];
  const float* rnn2_h     = (const float*)d_in[3];
  const float* osc_h      = (const float*)d_in[4];
  const float* init_w     = (const float*)d_in[5];
  const float* init_kappa = (const float*)d_in[6];
  const float* r1_jx_W = (const float*)d_in[7];  const float* r1_jx_b = (const float*)d_in[8];
  const float* r1_jh_W = (const float*)d_in[9];  const float* r1_jh_b = (const float*)d_in[10];
  const float* r1_kx_W = (const float*)d_in[11]; const float* r1_kx_b = (const float*)d_in[12];
  const float* r1_kh_W = (const float*)d_in[13]; const float* r1_kh_b = (const float*)d_in[14];
  const float* r2_jx_W = (const float*)d_in[15]; const float* r2_jx_b = (const float*)d_in[16];
  const float* r2_jh_W = (const float*)d_in[17]; const float* r2_jh_b = (const float*)d_in[18];
  const float* r2_kx_W = (const float*)d_in[19]; const float* r2_kx_b = (const float*)d_in[20];
  const float* r2_kh_W = (const float*)d_in[21]; const float* r2_kh_b = (const float*)d_in[22];
  const float* osc_jx_W = (const float*)d_in[23]; const float* osc_jx_b = (const float*)d_in[24];
  const float* osc_jh_W = (const float*)d_in[25]; const float* osc_jh_b = (const float*)d_in[26];
  const float* osc_kx_W = (const float*)d_in[27]; const float* osc_kx_b = (const float*)d_in[28];
  const float* osc_kh_W = (const float*)d_in[29]; const float* osc_kh_b = (const float*)d_in[30];
  const float* win_W = (const float*)d_in[31]; const float* win_b = (const float*)d_in[32];
  const float* lin_W = (const float*)d_in[33]; const float* lin_b = (const float*)d_in[34];
  float* out = (float*)d_out;

  // ---- workspace bump allocator ----
  char* base = (char*)d_ws;
  size_t off = 0;
  auto bump = [&](size_t bytes) -> void* {
    void* p = base + off;
    off = (off + bytes + 255) & ~(size_t)255;
    return p;
  };
  bf16_t* wt_r1j = (bf16_t*)bump(2ull * Sc * KP1);
  bf16_t* wt_r1k = (bf16_t*)bump(2ull * Sc * KP1);
  bf16_t* wt_r2j = (bf16_t*)bump(2ull * Sc * KP2);
  bf16_t* wt_r2k = (bf16_t*)bump(2ull * Sc * KP2);
  bf16_t* wt_osc[8];
  for (int i = 0; i < 8; ++i) wt_osc[i] = (bf16_t*)bump(2ull * Sc * KPO);
  bf16_t* wt_win = (bf16_t*)bump(2ull * NWIN * KPH);
  bf16_t* wt_lin = (bf16_t*)bump(2ull * NLIN * KPH);
  float* b_r1j = (float*)bump(4ull * Sc);
  float* b_r1k = (float*)bump(4ull * Sc);
  float* b_r2j = (float*)bump(4ull * Sc);
  float* b_r2k = (float*)bump(4ull * Sc);
  float* b_osc[8];
  for (int i = 0; i < 8; ++i) b_osc[i] = (float*)bump(4ull * Sc);
  float* b_win = (float*)bump(4ull * NWIN);
  float* b_lin = (float*)bump(4ull * NLIN);
  bf16_t* a_in1    = (bf16_t*)bump(2ull * Bc * KP1);
  bf16_t* a_in2    = (bf16_t*)bump(2ull * Bc * KP2);
  bf16_t* a_osc    = (bf16_t*)bump(2ull * Bc * KPO);
  bf16_t* h1_bf    = (bf16_t*)bump(2ull * Bc * Sc);
  bf16_t* a_oscout = (bf16_t*)bump(2ull * Rc * Sc);
  float* pre_j  = (float*)bump(4ull * Bc * Sc);
  float* pre_k  = (float*)bump(4ull * Bc * Sc);
  float* kg     = (float*)bump(4ull * Bc * NWIN);
  float* mdn    = (float*)bump(4ull * Rc * NLIN);

  const int SS = Sc * Sc;
  auto gemm = [&](const bf16_t* A, const bf16_t* BT, const float* bias, float* Cm,
                  int M, int Npad, int Kp) {
    int tiles = (M / 32) * (Npad / 64);
    gemm_bf16_wmma<<<dim3((tiles + 3) / 4), dim3(128), 0, stream>>>(A, BT, bias, Cm, M, Npad, Kp);
  };

  // ---- 1. pack fused transposed bf16 weights + combined biases ----
  pack_wt<<<dim3((KP1 + 255) / 256, Sc), 256, 0, stream>>>(wt_r1j, KP1, Sc, r1_jx_W, 131, r1_jh_W, Sc);
  pack_wt<<<dim3((KP1 + 255) / 256, Sc), 256, 0, stream>>>(wt_r1k, KP1, Sc, r1_kx_W, 131, r1_kh_W, Sc);
  pack_wt<<<dim3((KP2 + 255) / 256, Sc), 256, 0, stream>>>(wt_r2j, KP2, Sc, r2_jx_W, 643, r2_jh_W, Sc);
  pack_wt<<<dim3((KP2 + 255) / 256, Sc), 256, 0, stream>>>(wt_r2k, KP2, Sc, r2_kx_W, 643, r2_kh_W, Sc);
  for (int n = 0; n < 4; ++n) {
    pack_wt<<<dim3((KPO + 255) / 256, Sc), 256, 0, stream>>>(
        wt_osc[2 * n], KPO, Sc, osc_jx_W + (size_t)n * SS, Sc, osc_jh_W + (size_t)n * SS, Sc);
    pack_wt<<<dim3((KPO + 255) / 256, Sc), 256, 0, stream>>>(
        wt_osc[2 * n + 1], KPO, Sc, osc_kx_W + (size_t)n * SS, Sc, osc_kh_W + (size_t)n * SS, Sc);
  }
  pack_wt<<<dim3((KPH + 255) / 256, NWIN), 256, 0, stream>>>(wt_win, KPH, 60, win_W, Sc, nullptr, 0);
  pack_wt<<<dim3((KPH + 255) / 256, NLIN), 256, 0, stream>>>(wt_lin, KPH, 121, lin_W, Sc, nullptr, 0);
  pack_bias<<<2, 256, 0, stream>>>(b_r1j, r1_jx_b, r1_jh_b, Sc, Sc);
  pack_bias<<<2, 256, 0, stream>>>(b_r1k, r1_kx_b, r1_kh_b, Sc, Sc);
  pack_bias<<<2, 256, 0, stream>>>(b_r2j, r2_jx_b, r2_jh_b, Sc, Sc);
  pack_bias<<<2, 256, 0, stream>>>(b_r2k, r2_kx_b, r2_kh_b, Sc, Sc);
  for (int n = 0; n < 4; ++n) {
    pack_bias<<<2, 256, 0, stream>>>(b_osc[2 * n],     osc_jx_b + n * Sc, osc_jh_b + n * Sc, Sc, Sc);
    pack_bias<<<2, 256, 0, stream>>>(b_osc[2 * n + 1], osc_kx_b + n * Sc, osc_kh_b + n * Sc, Sc, Sc);
  }
  pack_bias<<<1, 256, 0, stream>>>(b_win, win_b, nullptr, 60, NWIN);
  pack_bias<<<1, 256, 0, stream>>>(b_lin, lin_b, nullptr, 121, NLIN);

  const int BS = Bc * Sc;
  // ---- 2. rnn1: in1 = [x | init_w | rnn1_h] -> h1 ----
  concat_bf16<<<dim3((KP1 + 255) / 256, Bc), 256, 0, stream>>>(
      a_in1, KP1, x, 3, init_w, Cc, rnn1_h, Sc, nullptr, 0);
  gemm(a_in1, wt_r1j, b_r1j, pre_j, Bc, Sc, KP1);
  gemm(a_in1, wt_r1k, b_r1k, pre_k, Bc, Sc, KP1);
  gate_combine<<<(BS + 255) / 256, 256, 0, stream>>>(pre_j, pre_k, rnn1_h,
                                                     out + OFF_H1, h1_bf, BS);
  // ---- 3. window attention ----
  gemm(h1_bf, wt_win, b_win, kg, Bc, NWIN, KPH);
  window_kernel<<<Bc, 256, 0, stream>>>(kg, init_kappa, out + OFF_PHI, out + OFF_KAPPA);
  attend_kernel<<<Bc, 128, 0, stream>>>(out + OFF_PHI, c_vec, out + OFF_W);
  // ---- 4. rnn2: in2 = [x | h1 | w | rnn2_h] -> h2 ----
  concat_bf16<<<dim3((KP2 + 255) / 256, Bc), 256, 0, stream>>>(
      a_in2, KP2, x, 3, out + OFF_H1, Sc, out + OFF_W, Cc, rnn2_h, Sc);
  gemm(a_in2, wt_r2j, b_r2j, pre_j, Bc, Sc, KP2);
  gemm(a_in2, wt_r2k, b_r2k, pre_k, Bc, Sc, KP2);
  gate_combine<<<(BS + 255) / 256, 256, 0, stream>>>(pre_j, pre_k, rnn2_h,
                                                     out + OFF_H2, nullptr, BS);
  // ---- 5. FFOsc: 4 gated cells with [h2 | osc_h[n]] ----
  for (int n = 0; n < 4; ++n) {
    concat_bf16<<<dim3((KPO + 255) / 256, Bc), 256, 0, stream>>>(
        a_osc, KPO, out + OFF_H2, Sc, osc_h + (size_t)n * BS, Sc, nullptr, 0, nullptr, 0);
    gemm(a_osc, wt_osc[2 * n],     b_osc[2 * n],     pre_j, Bc, Sc, KPO);
    gemm(a_osc, wt_osc[2 * n + 1], b_osc[2 * n + 1], pre_k, Bc, Sc, KPO);
    gate_combine<<<(BS + 255) / 256, 256, 0, stream>>>(
        pre_j, pre_k, osc_h + (size_t)n * BS, out + OFF_OSC + (size_t)n * BS, nullptr, BS);
  }
  // ---- 6. oscillator recurrence + MDN head ----
  osc_recur<<<(BS + 255) / 256, 256, 0, stream>>>(out + OFF_OSC, a_oscout);
  gemm(a_oscout, wt_lin, b_lin, mdn, Rc, NLIN, KPH);
  mdn_epilogue<<<(Rc + 127) / 128, 128, 0, stream>>>(mdn, out, Rc);
}
